// DecoderRNN_21569325761118
// MI455X (gfx1250) — matrix operand order repaired and checked by reading, not compiled, not run on hardware
//
#include <hip/hip_runtime.h>
#include <hip/hip_bf16.h>

// ---------------- problem constants ----------------
#define Bk   128
#define Nk   196
#define ENCk 512
#define Ek   256
#define Ak   256
#define Hk   512
#define Vk   512
#define Tk   512
#define XK   1280   // E + ENC + H  (fused gates GEMM K)
#define GJ   2048   // 4H
#define PJ   768    // V + A  (fused fcn + att_dec GEMM J)

typedef __attribute__((ext_vector_type(16))) __bf16 v16bf;
typedef __attribute__((ext_vector_type(8)))  __bf16 v8bf;
typedef __attribute__((ext_vector_type(8)))  float  v8f;

__device__ __forceinline__ __bf16 f2bf(float x) {
  unsigned u = __builtin_bit_cast(unsigned, x);
  unsigned r = (u + 0x7FFFu + ((u >> 16) & 1u)) >> 16;
  unsigned short s = (unsigned short)r;
  return __builtin_bit_cast(__bf16, s);
}
__device__ __forceinline__ float sigmoidf_(float x) { return 1.0f / (1.0f + __expf(-x)); }

// ---------------- WMMA tile helpers (bf16, f32 accum) ----------------
// A fragment: 16x32 bf16 row-major from X (leading dim ldx).
// Layout: lanes 0-15 hold row M=lane, K {k0..k0+15};
// lanes 16-31 hold row M=lane-16, K {k0+16..k0+31} (matches B fragment order).
__device__ __forceinline__ v16bf load_a_frag(const __bf16* __restrict__ X, int ldx,
                                             int m0, int k0, int lane) {
  int m  = m0 + (lane & 15);
  int kb = k0 + ((lane >> 4) << 4);
  const __bf16* p = X + (size_t)m * ldx + kb;
  union { v16bf v; v8bf h[2]; } u;
  u.h[0] = *(const v8bf*)(p);
  u.h[1] = *(const v8bf*)(p + 8);
  return u.v;
}

// B fragment: 32x16 bf16. Pre-packed: Bp[((kb*J + n)*32) + i] = W[kb*32 + i][n].
// Lane L (<16) needs K 0..15 of column n0+L; lane L(>=16) needs K 16..31 of column n0+L-16:
// 16 contiguous bf16 -> single 32-byte load.
__device__ __forceinline__ v16bf load_b_frag(const __bf16* __restrict__ Bp, int J,
                                             int kb, int n0, int lane) {
  const __bf16* p = Bp + (((size_t)kb * J + n0 + (lane & 15)) << 5) + ((lane >> 4) << 4);
  return *(const v16bf*)(p);
}

__device__ __forceinline__ v8f wmma_mn(const __bf16* __restrict__ X, int ldx,
                                       const __bf16* __restrict__ Bp, int J,
                                       int m0, int n0, int kblocks, v8f c, int lane) {
  for (int kb = 0; kb < kblocks; ++kb) {
    v16bf a = load_a_frag(X, ldx, m0, kb * 32, lane);
    v16bf b = load_b_frag(Bp, J, kb, n0, lane);
    c = __builtin_amdgcn_wmma_f32_16x16x32_bf16(false, a, false, b, (short)0, c, false, false);
  }
  return c;
}

// ---------------- one-time setup kernels ----------------
// Pack fp32 weight into bf16 B-fragment layout: dst[((kb_off+k/32)*Jdst + n_off + n)*32 + k%32]
// = src[k*k_stride + n*n_stride]
__global__ void pack_kernel(const float* __restrict__ src, __bf16* __restrict__ dst,
                            int Ksrc, int N, int Jdst, int n_off, int kb_off,
                            int k_stride, int n_stride) {
  long t = (long)blockIdx.x * blockDim.x + threadIdx.x;
  if (t >= (long)Ksrc * N) return;
  int k = (int)(t % Ksrc);
  int n = (int)(t / Ksrc);
  float v = src[(size_t)k * k_stride + (size_t)n * n_stride];
  int kb = kb_off + (k >> 5), i = k & 31;
  dst[(((size_t)kb * Jdst + n_off + n) << 5) + i] = f2bf(v);
}

__global__ void bias_kernel(const float* __restrict__ bih, const float* __restrict__ bhh,
                            const float* __restrict__ fcnb, const float* __restrict__ adecb,
                            float* __restrict__ gbias, float* __restrict__ pbias) {
  int t = blockIdx.x * blockDim.x + threadIdx.x;
  if (t < GJ) gbias[t] = bih[t] + bhh[t];
  if (t < PJ) pbias[t] = (t < Vk) ? fcnb[t] : adecb[t - Vk];
}

__global__ void feat2bf_kernel(const float* __restrict__ f, __bf16* __restrict__ fb, long n) {
  long t = (long)blockIdx.x * blockDim.x + threadIdx.x;
  if (t < n) fb[t] = f2bf(f[t]);
}

__global__ void mean_kernel(const float* __restrict__ features, float* __restrict__ mean) {
  int b = blockIdx.x;
  for (int d = threadIdx.x; d < ENCk; d += blockDim.x) {
    const float* p = features + (size_t)b * Nk * ENCk + d;
    float s = 0.f;
    for (int n = 0; n < Nk; ++n) s += p[(size_t)n * ENCk];
    mean[(size_t)b * ENCk + d] = s * (1.0f / Nk);
  }
}

// h0 = mean@init_h_W + b (stored bf16 into xh h-section); c0 = mean@init_c_W + b (f32)
__global__ void init_state_kernel(const float* __restrict__ mean,
                                  const float* __restrict__ ihW, const float* __restrict__ ihb,
                                  const float* __restrict__ icW, const float* __restrict__ icb,
                                  __bf16* __restrict__ xh, float* __restrict__ cbuf) {
  int t = blockIdx.x * blockDim.x + threadIdx.x;   // B*H threads
  int b = t >> 9, j = t & (Hk - 1);
  const float* m = mean + (size_t)b * ENCk;
  float ah = 0.f, ac = 0.f;
  for (int k = 0; k < ENCk; ++k) {
    float mv = m[k];
    ah += mv * ihW[(size_t)k * Hk + j];
    ac += mv * icW[(size_t)k * Hk + j];
  }
  xh[(size_t)b * XK + Ek + ENCk + j] = f2bf(ah + ihb[j]);
  cbuf[t] = ac + icb[j];
}

// ---------------- WMMA GEMM kernels ----------------
// enc_proj = features(bf16, 25088x512) @ att_enc_W(512x256) + b  -> f32
__global__ void __launch_bounds__(128) gemm_encproj(const __bf16* __restrict__ featbf,
                                                    const __bf16* __restrict__ We,
                                                    const float* __restrict__ encb,
                                                    float* __restrict__ encproj) {
  int lane = threadIdx.x & 31, wid = threadIdx.x >> 5;
  int n0 = blockIdx.x * 64 + wid * 16;
  int m0 = blockIdx.y * 16;
  float bv = encb[n0 + (lane & 15)];
  v8f c; for (int r = 0; r < 8; ++r) c[r] = bv;
  c = wmma_mn(featbf, ENCk, We, Ak, m0, n0, ENCk / 32, c, lane);
  int col = n0 + (lane & 15), rb = m0 + ((lane >> 4) << 3);
  for (int r = 0; r < 8; ++r) encproj[(size_t)(rb + r) * Ak + col] = c[r];
}

// gates = xh(128x1280 bf16) @ Wg(1280x2048) + gbias -> f32
__global__ void __launch_bounds__(128) gemm_gates(const __bf16* __restrict__ xh,
                                                  const __bf16* __restrict__ Wg,
                                                  const float* __restrict__ gbias,
                                                  float* __restrict__ gates) {
  int lane = threadIdx.x & 31, wid = threadIdx.x >> 5;
  int n0 = blockIdx.x * 64 + wid * 16;
  int m0 = blockIdx.y * 16;
  float bv = gbias[n0 + (lane & 15)];
  v8f c; for (int r = 0; r < 8; ++r) c[r] = bv;
  c = wmma_mn(xh, XK, Wg, GJ, m0, n0, XK / 32, c, lane);
  int col = n0 + (lane & 15), rb = m0 + ((lane >> 4) << 3);
  for (int r = 0; r < 8; ++r) gates[(size_t)(rb + r) * GJ + col] = c[r];
}

// [out | hproj] = h(bf16, in xh h-section) @ [fcn_W | att_dec_W](512x768) + [fcn_b|att_dec_b]
__global__ void __launch_bounds__(128) gemm_proj(const __bf16* __restrict__ xh,
                                                 const __bf16* __restrict__ Wp,
                                                 const float* __restrict__ pbias,
                                                 float* __restrict__ out,
                                                 float* __restrict__ hproj, int t) {
  int lane = threadIdx.x & 31, wid = threadIdx.x >> 5;
  int n0 = blockIdx.x * 64 + wid * 16;
  int m0 = blockIdx.y * 16;
  float bv = pbias[n0 + (lane & 15)];
  v8f c; for (int r = 0; r < 8; ++r) c[r] = bv;
  c = wmma_mn(xh + Ek + ENCk, XK, Wp, PJ, m0, n0, Hk / 32, c, lane);
  int col = n0 + (lane & 15), rb = m0 + ((lane >> 4) << 3);
  for (int r = 0; r < 8; ++r) {
    int row = rb + r;
    if (col < Vk) {
      if (t >= 0) out[(size_t)row * (Tk * Vk) + (size_t)t * Vk + col] = c[r];
    } else {
      hproj[(size_t)row * Ak + (col - Vk)] = c[r];
    }
  }
}

// ---------------- per-step scalar/VALU kernels ----------------
__device__ __forceinline__ float wave_sum(float v) {
  for (int o = 16; o; o >>= 1) v += __shfl_xor(v, o, 32);
  return v;
}

// attention: scores/softmax/context + embedding gather -> fills xh[:, 0:768] (bf16)
__global__ void __launch_bounds__(256) attn_kernel(const float* __restrict__ encproj,
                                                   const float* __restrict__ hproj,
                                                   const float* __restrict__ vW,
                                                   const float* __restrict__ vb,
                                                   const float* __restrict__ features,
                                                   const int* __restrict__ captions,
                                                   const float* __restrict__ embW,
                                                   __bf16* __restrict__ xh, int t) {
  __shared__ float hp[Ak];
  __shared__ float sc[Nk + 4];
  __shared__ float red[8];
  int b = blockIdx.x, tid = threadIdx.x;
  int lane = tid & 31, wid = tid >> 5;

  if (tid < Ak) hp[tid] = hproj[(size_t)b * Ak + tid];
  // embedding gather for this step
  int cap = captions[(size_t)b * (Tk + 1) + t];
  if (tid < Ek) xh[(size_t)b * XK + tid] = f2bf(embW[(size_t)cap * Ek + tid]);
  __syncthreads();

  const float* ep = encproj + (size_t)b * Nk * Ak;
  for (int n = wid; n < Nk; n += 8) {
    const float* row = ep + (size_t)n * Ak;
    float s = 0.f;
    for (int a = lane; a < Ak; a += 32) s += vW[a] * tanhf(row[a] + hp[a]);
    s = wave_sum(s);
    if (lane == 0) sc[n] = s + vb[0];
  }
  __syncthreads();

  // softmax over n
  float m = -3.4e38f;
  for (int n = tid; n < Nk; n += 256) m = fmaxf(m, sc[n]);
  for (int o = 16; o; o >>= 1) m = fmaxf(m, __shfl_xor(m, o, 32));
  if (lane == 0) red[wid] = m;
  __syncthreads();
  m = red[0];
  for (int i = 1; i < 8; ++i) m = fmaxf(m, red[i]);
  __syncthreads();

  float ps = 0.f;
  for (int n = tid; n < Nk; n += 256) { float e = __expf(sc[n] - m); sc[n] = e; ps += e; }
  ps = wave_sum(ps);
  if (lane == 0) red[wid] = ps;
  __syncthreads();
  float tot = 0.f;
  for (int i = 0; i < 8; ++i) tot += red[i];
  float inv = 1.0f / tot;
  __syncthreads();

  // context = sum_n alpha[n] * features[b,n,:]  (fp32 features for accuracy)
  const float* fb = features + (size_t)b * Nk * ENCk;
  for (int d = tid; d < ENCk; d += 256) {
    float acc = 0.f;
    for (int n = 0; n < Nk; ++n) acc += sc[n] * fb[(size_t)n * ENCk + d];
    xh[(size_t)b * XK + Ek + d] = f2bf(acc * inv);
  }
}

// LSTM cell: c_new = sig(f)*c + sig(i)*tanh(g); h_new = sig(o)*tanh(c_new) -> xh h-section (bf16)
__global__ void __launch_bounds__(256) cell_kernel(const float* __restrict__ gates,
                                                   float* __restrict__ cbuf,
                                                   __bf16* __restrict__ xh) {
  int t = blockIdx.x * blockDim.x + threadIdx.x;   // B*H
  int b = t >> 9, j = t & (Hk - 1);
  const float* g = gates + (size_t)b * GJ;
  float gi = sigmoidf_(g[j]);
  float gf = sigmoidf_(g[Hk + j]);
  float gg = tanhf(g[2 * Hk + j]);
  float go = sigmoidf_(g[3 * Hk + j]);
  float c = gf * cbuf[t] + gi * gg;
  cbuf[t] = c;
  xh[(size_t)b * XK + Ek + ENCk + j] = f2bf(go * tanhf(c));
}

// ---------------- launch ----------------
extern "C" void kernel_launch(void* const* d_in, const int* in_sizes, int n_in,
                              void* d_out, int out_size, void* d_ws, size_t ws_size,
                              hipStream_t stream) {
  const float* features = (const float*)d_in[0];
  const int*   captions = (const int*)d_in[1];
  const float* emb_W    = (const float*)d_in[2];
  const float* att_enc_W= (const float*)d_in[3];
  const float* att_enc_b= (const float*)d_in[4];
  const float* att_dec_W= (const float*)d_in[5];
  const float* att_dec_b= (const float*)d_in[6];
  const float* att_v_W  = (const float*)d_in[7];
  const float* att_v_b  = (const float*)d_in[8];
  const float* init_h_W = (const float*)d_in[9];
  const float* init_h_b = (const float*)d_in[10];
  const float* init_c_W = (const float*)d_in[11];
  const float* init_c_b = (const float*)d_in[12];
  const float* lstm_Wih = (const float*)d_in[13];
  const float* lstm_Whh = (const float*)d_in[14];
  const float* lstm_bih = (const float*)d_in[15];
  const float* lstm_bhh = (const float*)d_in[16];
  const float* fcn_W    = (const float*)d_in[17];
  const float* fcn_b    = (const float*)d_in[18];
  float* out = (float*)d_out;

  // workspace carve-out (256B aligned regions)
  char* ws = (char*)d_ws;
  size_t off = 0;
  auto carve = [&](size_t bytes) -> void* {
    void* p = ws + off;
    off = (off + bytes + 255) & ~(size_t)255;
    return p;
  };
  __bf16* Wg     = (__bf16*)carve((size_t)XK * GJ * 2);      // packed gates weights
  __bf16* Wp     = (__bf16*)carve((size_t)Hk * PJ * 2);      // packed [fcn|att_dec]
  __bf16* We     = (__bf16*)carve((size_t)ENCk * Ak * 2);    // packed att_enc
  __bf16* featbf = (__bf16*)carve((size_t)Bk * Nk * ENCk * 2);
  float*  encproj= (float*) carve((size_t)Bk * Nk * Ak * 4);
  __bf16* xh     = (__bf16*)carve((size_t)Bk * XK * 2);      // [emb|ctx|h] bf16
  float*  gates  = (float*) carve((size_t)Bk * GJ * 4);
  float*  cbuf   = (float*) carve((size_t)Bk * Hk * 4);
  float*  hproj  = (float*) carve((size_t)Bk * Ak * 4);
  float*  meanf  = (float*) carve((size_t)Bk * ENCk * 4);
  float*  gbias  = (float*) carve((size_t)GJ * 4);
  float*  pbias  = (float*) carve((size_t)PJ * 4);
  (void)ws_size; (void)in_sizes; (void)n_in; (void)out_size;

  // ---- one-time setup ----
  auto pk = [&](const float* src, __bf16* dst, int K, int N, int Jd, int noff, int kboff,
                int kst, int nst) {
    long tot = (long)K * N;
    pack_kernel<<<dim3((unsigned)((tot + 255) / 256)), 256, 0, stream>>>(
        src, dst, K, N, Jd, noff, kboff, kst, nst);
  };
  // Wg[k][j]: k<768 -> Wih[j][k] ; k>=768 -> Whh[j][k-768]
  pk(lstm_Wih, Wg, 768, GJ, GJ, 0, 0, 1, 768);
  pk(lstm_Whh, Wg, 512, GJ, GJ, 0, 24, 1, 512);
  // Wp[k][j]: j<512 -> fcn_W[k][j] ; j>=512 -> att_dec_W[k][j-512]
  pk(fcn_W,     Wp, Hk, Vk, PJ, 0,   0, Vk, 1);
  pk(att_dec_W, Wp, Hk, Ak, PJ, Vk,  0, Ak, 1);
  pk(att_enc_W, We, ENCk, Ak, Ak, 0, 0, Ak, 1);
  bias_kernel<<<8, 256, 0, stream>>>(lstm_bih, lstm_bhh, fcn_b, att_dec_b, gbias, pbias);

  long nf = (long)Bk * Nk * ENCk;
  feat2bf_kernel<<<dim3((unsigned)((nf + 255) / 256)), 256, 0, stream>>>(features, featbf, nf);
  mean_kernel<<<Bk, 256, 0, stream>>>(features, meanf);
  init_state_kernel<<<(Bk * Hk) / 256, 256, 0, stream>>>(meanf, init_h_W, init_h_b,
                                                         init_c_W, init_c_b, xh, cbuf);
  // enc_proj GEMM: M = B*N = 25088 rows (multiple of 16), J = 256
  gemm_encproj<<<dim3(Ak / 64, (Bk * Nk) / 16), 128, 0, stream>>>(featbf, We, att_enc_b, encproj);
  // hproj for t=0 from h0 (no output write)
  gemm_proj<<<dim3(PJ / 64, Bk / 16), 128, 0, stream>>>(xh, Wp, pbias, out, hproj, -1);

  // ---- recurrent loop ----
  for (int t = 0; t < Tk; ++t) {
    attn_kernel<<<Bk, 256, 0, stream>>>(encproj, hproj, att_v_W, att_v_b,
                                        features, captions, emb_W, xh, t);
    gemm_gates<<<dim3(GJ / 64, Bk / 16), 128, 0, stream>>>(xh, Wg, gbias, gates);
    cell_kernel<<<(Bk * Hk) / 256, 256, 0, stream>>>(gates, cbuf, xh);
    gemm_proj<<<dim3(PJ / 64, Bk / 16), 128, 0, stream>>>(xh, Wp, pbias, out, hproj, t);
  }
}
